// Head_549755814368
// MI455X (gfx1250) — matrix operand order repaired and checked by reading, not compiled, not run on hardware
//
#include <hip/hip_runtime.h>

// Shapes from the reference
#define Bb 4
#define Tt 4096
#define Cc 1024
#define Hh 128
#define BT (Bb * Tt)          // 16384

// Padded LDS pitches (elements); x2 bytes stays a multiple of 16 for b128.
#define KS_P 136   // K tile row pitch   (64 rows)
#define VS_P 72    // V^T tile row pitch (128 rows)
#define PS_P 72    // P scratch row pitch (16 rows per wave)
#define WT_P 1032  // weight slice pitch (16 rows of C=1024)

typedef __attribute__((ext_vector_type(16))) _Float16 v16h;
typedef __attribute__((ext_vector_type(8)))  float    v8f;

__device__ __forceinline__ v8f vzero8() {
    v8f z = {0.f, 0.f, 0.f, 0.f, 0.f, 0.f, 0.f, 0.f};
    return z;
}

// 16 contiguous f16 -> v16h via two b128 loads (p must be 16B aligned).
__device__ __forceinline__ v16h load_h16(const _Float16* p) {
    union { uint4 u[2]; v16h h; } c;
    c.u[0] = *(const uint4*)(p);
    c.u[1] = *(const uint4*)(p + 8);
    return c.h;
}

// A-fragment (16x32 f16): lane holds K = base..base+7 then 16+base..23+base.
__device__ __forceinline__ v16h load_afrag(const _Float16* p, int base) {
    union { uint4 u[2]; v16h h; } c;
    c.u[0] = *(const uint4*)(p + base);
    c.u[1] = *(const uint4*)(p + 16 + base);
    return c.h;
}

// CDNA5 async copy: global -> LDS, 16 bytes per lane, tracked by ASYNCcnt.
__device__ __forceinline__ void async_ld16(uint32_t lds_off, const void* g) {
    uint64_t ga = (uint64_t)(uintptr_t)g;
    asm volatile("global_load_async_to_lds_b128 %0, %1, off"
                 :: "v"(lds_off), "v"(ga) : "memory");
}
__device__ __forceinline__ void wait_async0() {
    asm volatile("s_wait_asynccnt 0x0" ::: "memory");
}
// Intra-wave LDS store->load ordering (P re-swizzle is wave-private).
__device__ __forceinline__ void wait_ds0() {
    asm volatile("s_wait_dscnt 0x0" ::: "memory");
}

// ---------------------------------------------------------------------------
// Kernel 1: fused projection  x[BT,C] @ {Wk,Wq,Wv}[C,H] -> K,Q (f16, [B,T,H])
// and V (f16, transposed [B,H,T] so attention's PV B-fragments are contiguous).
// grid = (BT/128, 24); block = 256 (8 waves); one 16x16 tile per wave.
// ---------------------------------------------------------------------------
__global__ __launch_bounds__(256, 1)
void proj_kernel(const float* __restrict__ x,
                 const float* __restrict__ Wk,
                 const float* __restrict__ Wq,
                 const float* __restrict__ Wv,
                 _Float16* __restrict__ Kb,
                 _Float16* __restrict__ Qb,
                 _Float16* __restrict__ Vt) {
    __shared__ __align__(16) _Float16 Wt[16 * WT_P];   // W[:, n0:n0+16]^T as f16

    const int mat = blockIdx.y >> 3;            // 0=K 1=Q 2=V
    const int n0  = (blockIdx.y & 7) * 16;
    const float* W = (mat == 0) ? Wk : (mat == 1) ? Wq : Wv;

    // Stage weight slice transposed: Wt[n][k] = W[k][n0+n].
    {
        const int t = threadIdx.x;
#pragma unroll
        for (int rr = 0; rr < 4; ++rr) {
            const int k = t * 4 + rr;
            const float4* src = (const float4*)(W + k * Hh + n0);
#pragma unroll
            for (int qq = 0; qq < 4; ++qq) {
                float4 w = src[qq];
                Wt[(qq * 4 + 0) * WT_P + k] = (_Float16)w.x;
                Wt[(qq * 4 + 1) * WT_P + k] = (_Float16)w.y;
                Wt[(qq * 4 + 2) * WT_P + k] = (_Float16)w.z;
                Wt[(qq * 4 + 3) * WT_P + k] = (_Float16)w.w;
            }
        }
    }
    __syncthreads();

    const int wave = threadIdx.x >> 5;
    const int lane = threadIdx.x & 31;
    const int half = (lane & 16) ? 1 : 0;

    const int rowTile = blockIdx.x * 128 + wave * 16;
    const int arow    = rowTile + (lane & 15);
    const int abase   = half ? 8 : 0;
    const int n       = lane & 15;
    const int kb16    = half ? 16 : 0;

    v8f acc = vzero8();

    for (int k0 = 0; k0 < Cc; k0 += 32) {
        // A fragment from fp32 x (b128 loads + v_cvt_pk_f16_f32).
        const float* xp = x + arow * Cc + k0;
        v16h a;
#pragma unroll
        for (int j = 0; j < 8; ++j) {
            a[j]     = (_Float16)xp[abase + j];
            a[8 + j] = (_Float16)xp[16 + abase + j];
        }
        // B fragment: contiguous run in transposed weight slice.
        v16h b = load_afrag(&Wt[n * WT_P + k0], abase);

        acc = __builtin_amdgcn_wmma_f32_16x16x32_f16(false, a, false, b,
                                                     (short)0, acc, false, false);
    }

    // C/D layout: element v -> row = v + 8*half, col = n.
#pragma unroll
    for (int v = 0; v < 8; ++v) {
        const int r = rowTile + v + half * 8;
        const _Float16 val = (_Float16)acc[v];
        if (mat == 2) {
            const int bb = r >> 12;            // r / T
            const int tt = r & (Tt - 1);       // r % T
            Vt[((size_t)bb * Hh + n0 + n) * Tt + tt] = val;
        } else {
            _Float16* Out = (mat == 0) ? Kb : Qb;
            Out[(size_t)r * Hh + n0 + n] = val;
        }
    }
}

// ---------------------------------------------------------------------------
// Kernel 2: causal flash attention, f16 WMMA with f32 accumulation.
// grid = (T/128, B); block = 256 (8 waves); wave w owns 16 query rows.
// K/V tiles double-buffered in LDS via global_load_async_to_lds_b128, so the
// ASYNCcnt DMA of tile i+1 overlaps the WMMA work on tile i.
// ---------------------------------------------------------------------------
__global__ __launch_bounds__(256, 1)
void attn_kernel(const _Float16* __restrict__ Qb,
                 const _Float16* __restrict__ Kb,
                 const _Float16* __restrict__ Vt,
                 float* __restrict__ out) {
    __shared__ __align__(16) _Float16 Ks[2][64 * KS_P];     // 2 x 17 KB, [key][h]
    __shared__ __align__(16) _Float16 Vs[2][128 * VS_P];    // 2 x 18 KB, [h][key]
    __shared__ __align__(16) _Float16 Ps[8 * 16 * PS_P];    // 18 KB, per-wave P

    const int bidx = blockIdx.y;
    const int q0b  = blockIdx.x * 128;
    const int t    = threadIdx.x;
    const int wave = t >> 5;
    const int lane = t & 31;
    const int half = (lane & 16) ? 1 : 0;

    const int q0    = q0b + wave * 16;
    const int abase = half ? 8 : 0;
    const int n     = lane & 15;
    const int kb16  = half ? 16 : 0;
    const int moff  = half * 8;

    const float scale = 0.03125f;  // C^-0.5 = 1/32

    // Q fragments (16 rows x 128), pinned in registers.
    v16h qf[4];
    {
        const int qrow = q0 + (lane & 15);
        const _Float16* qp = Qb + ((size_t)bidx * Tt + qrow) * Hh;
#pragma unroll
        for (int c = 0; c < 4; ++c) qf[c] = load_afrag(qp + c * 32, abase);
    }

    v8f acc[8];
#pragma unroll
    for (int h = 0; h < 8; ++h) acc[h] = vzero8();
    float mrow[8], lrow[8];
#pragma unroll
    for (int v = 0; v < 8; ++v) { mrow[v] = -INFINITY; lrow[v] = 0.f; }

    _Float16* Pw = Ps + wave * (16 * PS_P);
    const _Float16* Kbase = Kb + (size_t)bidx * Tt * Hh;   // [key][h]
    const _Float16* Vbase = Vt + (size_t)bidx * Hh * Tt;   // [h][key]
    uint32_t ksB[2] = { (uint32_t)(uintptr_t)&Ks[0][0], (uint32_t)(uintptr_t)&Ks[1][0] };
    uint32_t vsB[2] = { (uint32_t)(uintptr_t)&Vs[0][0], (uint32_t)(uintptr_t)&Vs[1][0] };

    const int jend = q0b + 128;  // block-uniform causal frontier

    // Async staging of one 64-key tile pair into buffer `p`.
    auto stage = [&](int j0, int p) {
        const _Float16* Kg = Kbase + (size_t)j0 * Hh;
#pragma unroll
        for (int i = 0; i < 4; ++i) {
            const int u = t + i * 256, r = u >> 4, c16 = u & 15;
            async_ld16(ksB[p] + (uint32_t)(r * KS_P + c16 * 8) * 2, Kg + u * 8);
        }
        const _Float16* Vg = Vbase + j0;
#pragma unroll
        for (int i = 0; i < 4; ++i) {
            const int u = t + i * 256, r = u >> 3, c8 = u & 7;
            async_ld16(vsB[p] + (uint32_t)(r * VS_P + c8 * 8) * 2,
                       Vg + (size_t)r * Tt + c8 * 8);
        }
    };

    stage(0, 0);  // prefetch first tile

    int p = 0;
    for (int j0 = 0; j0 < jend; j0 += 64, p ^= 1) {
        wait_async0();     // tile j0 resident in buffer p
        __syncthreads();   // all waves: tile ready, prev buffer's readers done

        if (j0 + 64 < jend) stage(j0 + 64, p ^ 1);  // DMA next tile during compute

        const _Float16* Ksp = &Ks[p][0];
        const _Float16* Vsp = &Vs[p][0];

        // S = Q K^T : 4 key sub-tiles, 4 chunks of K=32 each.
        v8f s[4];
#pragma unroll
        for (int kt = 0; kt < 4; ++kt) {
            s[kt] = vzero8();
            const _Float16* krow = Ksp + (kt * 16 + n) * KS_P;
#pragma unroll
            for (int c = 0; c < 4; ++c) {
                v16h bf = load_afrag(krow + c * 32, abase);
                s[kt] = __builtin_amdgcn_wmma_f32_16x16x32_f16(
                    false, qf[c], false, bf, (short)0, s[kt], false, false);
            }
        }

        // Scale + causal mask.
#pragma unroll
        for (int kt = 0; kt < 4; ++kt)
#pragma unroll
            for (int v = 0; v < 8; ++v) {
                const int key = j0 + kt * 16 + n;
                const int qr  = q0 + v + moff;
                const float sv = s[kt][v] * scale;
                s[kt][v] = (key <= qr) ? sv : -INFINITY;
            }

        // Online softmax: reductions stay inside each 16-lane half.
        float mt[8];
#pragma unroll
        for (int v = 0; v < 8; ++v)
            mt[v] = fmaxf(fmaxf(s[0][v], s[1][v]), fmaxf(s[2][v], s[3][v]));
#pragma unroll
        for (int off = 1; off <= 8; off <<= 1)
#pragma unroll
            for (int v = 0; v < 8; ++v)
                mt[v] = fmaxf(mt[v], __shfl_xor(mt[v], off, 32));

        float corr[8], rs[8];
#pragma unroll
        for (int v = 0; v < 8; ++v) {
            const float mn = fmaxf(mrow[v], mt[v]);
            corr[v] = __expf(mrow[v] - mn);
            mrow[v] = mn;
            rs[v]   = 0.f;
        }
#pragma unroll
        for (int kt = 0; kt < 4; ++kt)
#pragma unroll
            for (int v = 0; v < 8; ++v) {
                const float pe = __expf(s[kt][v] - mrow[v]);
                s[kt][v] = pe;
                rs[v] += pe;
            }
#pragma unroll
        for (int off = 1; off <= 8; off <<= 1)
#pragma unroll
            for (int v = 0; v < 8; ++v)
                rs[v] += __shfl_xor(rs[v], off, 32);
#pragma unroll
        for (int v = 0; v < 8; ++v) lrow[v] = lrow[v] * corr[v] + rs[v];
#pragma unroll
        for (int h = 0; h < 8; ++h)
#pragma unroll
            for (int v = 0; v < 8; ++v) acc[h][v] *= corr[v];

        // Re-swizzle P through wave-private LDS: C-layout -> A-layout.
        // Only intra-wave ordering needed -> s_wait_dscnt, no block barrier.
#pragma unroll
        for (int kt = 0; kt < 4; ++kt)
#pragma unroll
            for (int v = 0; v < 8; ++v)
                Pw[(v + moff) * PS_P + kt * 16 + n] = (_Float16)s[kt][v];
        wait_ds0();

        v16h pa[2];
#pragma unroll
        for (int c2 = 0; c2 < 2; ++c2)
            pa[c2] = load_afrag(Pw + (lane & 15) * PS_P + c2 * 32, abase);

        // acc += P @ V : contiguous b128 B-fragments from V^T tile.
#pragma unroll
        for (int h = 0; h < 8; ++h) {
            const _Float16* vrow = Vsp + (h * 16 + n) * VS_P;
#pragma unroll
            for (int c2 = 0; c2 < 2; ++c2) {
                v16h vf = load_afrag(vrow + c2 * 32, abase);
                acc[h] = __builtin_amdgcn_wmma_f32_16x16x32_f16(
                    false, pa[c2], false, vf, (short)0, acc[h], false, false);
            }
        }
    }

    // Normalize and store fp32 output [B,T,H].
#pragma unroll
    for (int h = 0; h < 8; ++h)
#pragma unroll
        for (int v = 0; v < 8; ++v) {
            const int qr = q0 + v + moff;
            out[((size_t)bidx * Tt + qr) * Hh + h * 16 + n] = acc[h][v] / lrow[v];
        }
}

// ---------------------------------------------------------------------------
// Launcher. Workspace (f16): K[B,T,H] | Q[B,T,H] | V^T[B,H,T], 4 MB each.
// ---------------------------------------------------------------------------
extern "C" void kernel_launch(void* const* d_in, const int* in_sizes, int n_in,
                              void* d_out, int out_size, void* d_ws, size_t ws_size,
                              hipStream_t stream) {
    const float* x  = (const float*)d_in[0];
    const float* Wk = (const float*)d_in[1];
    const float* Wq = (const float*)d_in[2];
    const float* Wv = (const float*)d_in[3];
    float* out = (float*)d_out;

    _Float16* Kb = (_Float16*)d_ws;
    _Float16* Qb = Kb + (size_t)BT * Hh;
    _Float16* Vt = Qb + (size_t)BT * Hh;

    dim3 block(256);
    dim3 grid1(BT / 128, 24);
    proj_kernel<<<grid1, block, 0, stream>>>(x, Wk, Wq, Wv, Kb, Qb, Vt);

    dim3 grid2(Tt / 128, Bb);
    attn_kernel<<<grid2, block, 0, stream>>>(Qb, Kb, Vt, out);
}
